// RadonSelfAttention_19267223290563
// MI455X (gfx1250) — matrix operand order repaired and checked by reading, not compiled, not run on hardware
//
#include <hip/hip_runtime.h>
#include <hip/hip_bf16.h>

#define PI_F 3.14159265358979323846f

typedef __attribute__((ext_vector_type(16))) _Float16 v16h;
typedef __attribute__((ext_vector_type(8)))  float    v8f;
typedef __attribute__((ext_vector_type(8)))  _Float16 h8;

// ---------------------------------------------------------------- constants
// BC=16 images, H=W=S=128, A=E=180 (pad 192), 3E=540 (pad 544), heads=4,
// dh=45 (pad 64 so K is a multiple of the 32-wide WMMA step)

// ---------------------------------------------------------------- trig table
__global__ void k_trig(float* trig) {
    int a = blockIdx.x * blockDim.x + threadIdx.x;
    if (a < 180) {
        float th = (float)a * (PI_F / 180.0f);
        trig[2 * a]     = cosf(th);
        trig[2 * a + 1] = sinf(th);
    }
}

// ---------------------------------------------------------------- radon
// grid (16, 12), block 128.  Image cached in LDS (64 KB).
__global__ void k_radon(const float* __restrict__ x, const float* __restrict__ trig,
                        float* __restrict__ sino) {
    __shared__ float img[128 * 128];
    const int b = blockIdx.x, chunk = blockIdx.y, tid = threadIdx.x;
    const float* src = x + (size_t)b * 16384;
    for (int i = tid; i < 16384; i += 128) img[i] = src[i];
    __syncthreads();

    const float c = 63.5f;
    const int d = tid;              // detector / output column
    const float xs = (float)d - c;
    for (int al = 0; al < 15; ++al) {
        int a = chunk * 15 + al;
        float co = trig[2 * a], si = trig[2 * a + 1];
        float sum = 0.0f;
        for (int y = 0; y < 128; ++y) {
            float ysv = (float)y - c;
            float sxf = co * xs + si * ysv + c;
            float syf = -si * xs + co * ysv + c;
            float y0f = floorf(syf), x0f = floorf(sxf);
            float wy = syf - y0f, wx = sxf - x0f;
            int iy = (int)y0f, ix = (int)x0f;
            float v00 = 0.f, v01 = 0.f, v10 = 0.f, v11 = 0.f;
            bool y0ok = (iy >= 0) & (iy < 128);
            bool y1ok = (iy + 1 >= 0) & (iy + 1 < 128);
            bool x0ok = (ix >= 0) & (ix < 128);
            bool x1ok = (ix + 1 >= 0) & (ix + 1 < 128);
            if (y0ok && x0ok) v00 = img[iy * 128 + ix];
            if (y0ok && x1ok) v01 = img[iy * 128 + ix + 1];
            if (y1ok && x0ok) v10 = img[(iy + 1) * 128 + ix];
            if (y1ok && x1ok) v11 = img[(iy + 1) * 128 + ix + 1];
            sum += (1.f - wy) * (1.f - wx) * v00 + (1.f - wy) * wx * v01
                 + wy * (1.f - wx) * v10 + wy * wx * v11;
        }
        sino[((size_t)b * 128 + d) * 180 + a] = sum;
    }
}

// ---------------------------------------------------------------- generic WMMA GEMM
// C[M,N] = scale * (A[M,K] @ Bt^T) + bias[n], where Bt is stored N-major:
// element B[k][n] == Bt[n*ldb + k].  f16 in, f32 accumulate.
// One wave per 16x16 tile; K compile-time multiple of 32 -> fully unrolled.
template <int K>
__global__ void k_wmma_gemm(const _Float16* __restrict__ A, int lda, int sA,
                            const _Float16* __restrict__ Bt, int ldb, int sB,
                            float* __restrict__ C, int ldc, int sC,
                            const float* __restrict__ bias, float scale) {
    const int mt = blockIdx.x, nt = blockIdx.y, bz = blockIdx.z;
    const int lane  = threadIdx.x;
    const int halfi = lane >> 4;     // 0 | 1
    const int mrow  = lane & 15;
    const int ncol  = lane & 15;
    const int nglob = nt * 16 + ncol;

    const _Float16* arow = A + (size_t)bz * sA + ((size_t)mt * 16 + mrow) * lda
                             + 8 * halfi;
    const _Float16* brow = Bt + (size_t)bz * sB + (size_t)nglob * ldb
                              + 8 * halfi;

    v8f acc = {};
#pragma unroll
    for (int k0 = 0; k0 < K; k0 += 32) {
        // A 16x32 f16 layout: lane-half h, element j: j<8 -> K=k0+8h+j,
        //                                            j>=8 -> K=k0+16+8h+(j-8)
        h8 a0 = *(const h8*)(arow + k0);
        h8 a1 = *(const h8*)(arow + k0 + 16);
        // B 32x16 f16 layout: same as A: j<8 -> K=k0+8h+j, j>=8 -> K=k0+16+8h+(j-8)
        h8 b0 = *(const h8*)(brow + k0);
        h8 b1 = *(const h8*)(brow + k0 + 16);
        v16h a = __builtin_shufflevector(a0, a1, 0, 1, 2, 3, 4, 5, 6, 7,
                                         8, 9, 10, 11, 12, 13, 14, 15);
        v16h b = __builtin_shufflevector(b0, b1, 0, 1, 2, 3, 4, 5, 6, 7,
                                         8, 9, 10, 11, 12, 13, 14, 15);
        acc = __builtin_amdgcn_wmma_f32_16x16x32_f16(false, a, false, b,
                                                     (short)0, acc, false, false);
    }
    float* Cb = C + (size_t)bz * sC;
    float bi = bias ? bias[nglob] : 0.0f;
#pragma unroll
    for (int r = 0; r < 8; ++r) {
        int mglob = mt * 16 + r + 8 * halfi;
        Cb[(size_t)mglob * ldc + nglob] = acc[r] * scale + bi;
    }
}

// ---------------------------------------------------------------- pack / pad kernels
__global__ void k_pack_xh(const float* __restrict__ sino, _Float16* __restrict__ Xh) {
    int t = blockIdx.x * blockDim.x + threadIdx.x;
    if (t >= 16 * 128 * 192) return;
    int k = t % 192, s = t / 192;
    Xh[t] = (k < 180) ? (_Float16)sino[(size_t)s * 180 + k] : (_Float16)0.0f;
}

// in_proj_w is (540,180) = [n][k]; Bt wants [n][k] -> pad-copy to (544,192)
__global__ void k_pack_wqkv(const float* __restrict__ w, _Float16* __restrict__ Wh) {
    int t = blockIdx.x * blockDim.x + threadIdx.x;
    if (t >= 544 * 192) return;
    int k = t % 192, n = t / 192;
    Wh[t] = (n < 540 && k < 180) ? (_Float16)w[(size_t)n * 180 + k] : (_Float16)0.0f;
}

__global__ void k_pack_biasq(const float* __restrict__ b, float* __restrict__ bq) {
    int t = blockIdx.x * blockDim.x + threadIdx.x;
    if (t < 544) bq[t] = (t < 540) ? b[t] : 0.0f;
}

// Qh/Kh: (b,h, s, d64) row-major.  V packed transposed: Vt (b,h, d64, s).
__global__ void k_split_heads(const float* __restrict__ QKV, _Float16* __restrict__ Qh,
                              _Float16* __restrict__ Kh, _Float16* __restrict__ Vt) {
    int t = blockIdx.x * blockDim.x + threadIdx.x;
    const int per = 16 * 4 * 128 * 64;
    if (t >= 3 * per) return;
    int which = t / per, r = t % per;
    int d = r % 64, s = (r / 64) % 128, h = (r / (64 * 128)) % 4, b = r / (64 * 128 * 4);
    _Float16 v = (_Float16)0.0f;
    if (d < 45)
        v = (_Float16)QKV[((size_t)b * 128 + s) * 544 + which * 180 + h * 45 + d];
    if (which == 0)
        Qh[(((size_t)b * 4 + h) * 128 + s) * 64 + d] = v;
    else if (which == 1)
        Kh[(((size_t)b * 4 + h) * 128 + s) * 64 + d] = v;
    else
        Vt[(((size_t)b * 4 + h) * 64 + d) * 128 + s] = v;
}

// ---------------------------------------------------------------- softmax (rows of 128)
__global__ void k_softmax(const float* __restrict__ S, _Float16* __restrict__ P) {
    __shared__ float buf[128];
    const int row = blockIdx.x, tid = threadIdx.x;
    float v = S[(size_t)row * 128 + tid];
    buf[tid] = v; __syncthreads();
    for (int s = 64; s > 0; s >>= 1) {
        if (tid < s) buf[tid] = fmaxf(buf[tid], buf[tid + s]);
        __syncthreads();
    }
    float m = buf[0]; __syncthreads();
    float e = expf(v - m);
    buf[tid] = e; __syncthreads();
    for (int s = 64; s > 0; s >>= 1) {
        if (tid < s) buf[tid] += buf[tid + s];
        __syncthreads();
    }
    P[(size_t)row * 128 + tid] = (_Float16)(e / buf[0]);
}

__global__ void k_gather_o(const float* __restrict__ O, _Float16* __restrict__ Oeh) {
    int t = blockIdx.x * blockDim.x + threadIdx.x;
    if (t >= 16 * 128 * 192) return;
    int e = t % 192, s = (t / 192) % 128, b = t / (192 * 128);
    _Float16 v = (_Float16)0.0f;
    if (e < 180) {
        int h = e / 45, d = e % 45;
        v = (_Float16)O[(((size_t)b * 4 + h) * 128 + s) * 64 + d];
    }
    Oeh[t] = v;
}

// out_proj_w is (180,180) = [n][e]; Bt wants [n][k=e] -> pad-copy to (192,192)
__global__ void k_pack_wo(const float* __restrict__ w, _Float16* __restrict__ Wh) {
    int t = blockIdx.x * blockDim.x + threadIdx.x;
    if (t >= 192 * 192) return;
    int k = t % 192, n = t / 192;
    Wh[t] = (n < 180 && k < 180) ? (_Float16)w[(size_t)n * 180 + k] : (_Float16)0.0f;
}

__global__ void k_pack_biaso(const float* __restrict__ b, float* __restrict__ bo) {
    int t = blockIdx.x * blockDim.x + threadIdx.x;
    if (t < 192) bo[t] = (t < 180) ? b[t] : 0.0f;
}

// Closed-form Ram-Lak circular-convolution matrix (exact FFT-filter equivalent).
__global__ void k_build_g(_Float16* __restrict__ G) {
    int t = blockIdx.x * blockDim.x + threadIdx.x;
    if (t >= 128 * 128) return;
    int d = t / 128, k = t % 128, m = d - k;
    float v = 0.0f;
    if (m == 0) v = 0.5f;
    else if (m & 1) v = -2.0f / (PI_F * PI_F * (float)(m * m));
    G[t] = (_Float16)v;
}

// Attention output Y (b, det, 192) -> transposed f16 Yt (b, angle192, det128)
__global__ void k_cvt_yt(const float* __restrict__ Y, _Float16* __restrict__ Yt) {
    int t = blockIdx.x * blockDim.x + threadIdx.x;
    if (t >= 16 * 192 * 128) return;
    int det = t % 128, a = (t / 128) % 192, b = t / (128 * 192);
    Yt[t] = (_Float16)Y[((size_t)b * 128 + det) * 192 + a];
}

// ---------------------------------------------------------------- backprojection
__global__ void k_backproj(const float* __restrict__ F, const float* __restrict__ trig,
                           float* __restrict__ out) {
    const int b = blockIdx.x;
    const int p = blockIdx.y * 256 + threadIdx.x;        // 0..16383
    const int y = p >> 7, x = p & 127;
    const float c = 63.5f;
    const float ys = (float)y - c, xs = (float)x - c;
    const float* Fb = F + (size_t)b * 128 * 192;
    float acc = 0.0f;
    for (int a = 0; a < 180; ++a) {
        float co = trig[2 * a], si = trig[2 * a + 1];
        float t = ys * co - xs * si + c;
        float t0 = floorf(t);
        float w = t - t0;
        int i0 = (int)t0; i0 = i0 < 0 ? 0 : (i0 > 127 ? 127 : i0);
        int i1 = i0 + 1 > 127 ? 127 : i0 + 1;
        float v = (1.0f - w) * Fb[(size_t)i0 * 192 + a] + w * Fb[(size_t)i1 * 192 + a];
        acc += (t >= 0.0f && t <= 127.0f) ? v : 0.0f;
    }
    float res = acc * (PI_F / 360.0f);
    if (xs * xs + ys * ys > c * c) res = 0.0f;
    out[(size_t)b * 16384 + p] = res;
}

// ---------------------------------------------------------------- host launcher
extern "C" void kernel_launch(void* const* d_in, const int* in_sizes, int n_in,
                              void* d_out, int out_size, void* d_ws, size_t ws_size,
                              hipStream_t stream) {
    const float* x     = (const float*)d_in[0];   // (8,2,128,128)
    const float* in_w  = (const float*)d_in[1];   // (540,180)
    const float* in_b  = (const float*)d_in[2];   // (540,)
    const float* out_w = (const float*)d_in[3];   // (180,180)
    const float* out_b = (const float*)d_in[4];   // (180,)
    float* out = (float*)d_out;

    char* ws = (char*)d_ws;
    size_t off = 0;
    auto alloc = [&](size_t bytes) -> char* {
        char* p = ws + off;
        off = (off + bytes + 255) & ~(size_t)255;
        return p;
    };
    float*     trig  = (float*)    alloc((size_t)180 * 2 * 4);
    float*     sino  = (float*)    alloc((size_t)16 * 128 * 180 * 4);
    _Float16*  Xh    = (_Float16*) alloc((size_t)16 * 128 * 192 * 2);
    _Float16*  Wqkv  = (_Float16*) alloc((size_t)544 * 192 * 2);
    float*     biasQ = (float*)    alloc((size_t)544 * 4);
    float*     QKV   = (float*)    alloc((size_t)16 * 128 * 544 * 4);
    _Float16*  Qh    = (_Float16*) alloc((size_t)16 * 4 * 128 * 64 * 2);
    _Float16*  Kh    = (_Float16*) alloc((size_t)16 * 4 * 128 * 64 * 2);
    _Float16*  Vt    = (_Float16*) alloc((size_t)16 * 4 * 64 * 128 * 2);
    float*     Sc    = (float*)    alloc((size_t)16 * 4 * 128 * 128 * 4);
    _Float16*  Ph    = (_Float16*) alloc((size_t)16 * 4 * 128 * 128 * 2);
    float*     Obuf  = (float*)    alloc((size_t)16 * 4 * 128 * 64 * 4);
    _Float16*  Oeh   = (_Float16*) alloc((size_t)16 * 128 * 192 * 2);
    _Float16*  Woh   = (_Float16*) alloc((size_t)192 * 192 * 2);
    float*     biasO = (float*)    alloc((size_t)192 * 4);
    float*     Y     = (float*)    alloc((size_t)16 * 128 * 192 * 4);
    _Float16*  Gh    = (_Float16*) alloc((size_t)128 * 128 * 2);
    _Float16*  Yt    = (_Float16*) alloc((size_t)16 * 192 * 128 * 2);
    float*     F     = (float*)    alloc((size_t)16 * 128 * 192 * 4);

    // 1) angle tables + radon
    k_trig<<<1, 256, 0, stream>>>(trig);
    k_radon<<<dim3(16, 12), 128, 0, stream>>>(x, trig, sino);

    // 2) QKV projection: (128 x 192) @ (192 x 544) per image
    k_pack_xh  <<<(16 * 128 * 192 + 255) / 256, 256, 0, stream>>>(sino, Xh);
    k_pack_wqkv<<<(544 * 192 + 255) / 256,      256, 0, stream>>>(in_w, Wqkv);
    k_pack_biasq<<<3, 256, 0, stream>>>(in_b, biasQ);
    k_wmma_gemm<192><<<dim3(8, 34, 16), 32, 0, stream>>>(
        Xh, 192, 128 * 192, Wqkv, 192, 0, QKV, 544, 128 * 544, biasQ, 1.0f);

    // 3) attention: scores = Q K^T / sqrt(45); softmax; O = P V
    k_split_heads<<<(3 * 16 * 4 * 128 * 64 + 255) / 256, 256, 0, stream>>>(QKV, Qh, Kh, Vt);
    k_wmma_gemm<64><<<dim3(8, 8, 64), 32, 0, stream>>>(
        Qh, 64, 128 * 64, Kh, 64, 128 * 64, Sc, 128, 128 * 128,
        nullptr, 0.14907119849998599f);
    k_softmax<<<16 * 4 * 128, 128, 0, stream>>>(Sc, Ph);
    k_wmma_gemm<128><<<dim3(8, 4, 64), 32, 0, stream>>>(
        Ph, 128, 128 * 128, Vt, 128, 64 * 128, Obuf, 64, 128 * 64,
        nullptr, 1.0f);

    // 4) output projection: (128 x 192) @ (192 x 192) per image
    k_gather_o<<<(16 * 128 * 192 + 255) / 256, 256, 0, stream>>>(Obuf, Oeh);
    k_pack_wo <<<(192 * 192 + 255) / 256,      256, 0, stream>>>(out_w, Woh);
    k_pack_biaso<<<1, 256, 0, stream>>>(out_b, biasO);
    k_wmma_gemm<192><<<dim3(8, 12, 16), 32, 0, stream>>>(
        Oeh, 192, 128 * 192, Woh, 192, 0, Y, 192, 128 * 192, biasO, 1.0f);

    // 5) ramp filter as closed-form circular-convolution GEMM (exact FFT equivalent)
    k_build_g<<<64, 256, 0, stream>>>(Gh);
    k_cvt_yt <<<(16 * 192 * 128 + 255) / 256, 256, 0, stream>>>(Y, Yt);
    k_wmma_gemm<128><<<dim3(8, 12, 16), 32, 0, stream>>>(
        Gh, 128, 0, Yt, 128, 192 * 128, F, 192, 128 * 192, nullptr, 1.0f);

    // 6) filtered backprojection
    k_backproj<<<dim3(16, 64), 256, 0, stream>>>(F, trig, out);
}
